// OHiAttention_1649267441670
// MI455X (gfx1250) — matrix operand order repaired and checked by reading, not compiled, not run on hardware
//
#include <hip/hip_runtime.h>

// Problem constants (reference: HIDDEN=512, N_HEAD=8, D_HEAD=64, MAX_REL=3, B=2, S=2048)
#define Bb   2
#define SEQ  2048
#define HID  512
#define NH   8
#define DH   64

typedef __attribute__((ext_vector_type(16))) __bf16 v16bf;
typedef __attribute__((ext_vector_type(8)))  __bf16 v8bf;
typedef __attribute__((ext_vector_type(4)))  __bf16 v4bf;
typedef __attribute__((ext_vector_type(8)))  float  v8f;

// ---------------------------------------------------------------------------
// WMMA fragment loaders per CDNA5 ISA 7.12.2 (wave32):
//  A 16x32 bf16: lane m=lane&15, half=lane>>4; t<8 -> K=half*8+t, t>=8 -> K=16+half*8+(t-8)
//    => two contiguous 8-element chunks at +0 and +16 from (row, half*8).
//  B 32x16 bf16: lane n=lane&15, half=lane>>4; same K order as A
//    => two contiguous 8-element chunks at +0 and +16 from (col, half*8).
// ---------------------------------------------------------------------------
static __device__ __forceinline__ v16bf ld_a_frag(const __bf16* p) {
  v8bf lo = *(const v8bf*)(p);
  v8bf hi = *(const v8bf*)(p + 16);
  v16bf r;
#pragma unroll
  for (int t = 0; t < 8; ++t) { r[t] = lo[t]; r[t + 8] = hi[t]; }
  return r;
}

static __device__ __forceinline__ v16bf ld_b_frag(const __bf16* p) {
  v8bf lo = *(const v8bf*)(p);
  v8bf hi = *(const v8bf*)(p + 16);
  v16bf r;
#pragma unroll
  for (int t = 0; t < 8; ++t) { r[t] = lo[t]; r[t + 8] = hi[t]; }
  return r;
}

static __device__ __forceinline__ v8f wmma_bf16(v16bf a, v16bf b, v8f c) {
  // D = A(16x32) * B(32x16) + C, fp32 accumulate
  return __builtin_amdgcn_wmma_f32_16x16x32_bf16(false, a, false, b, (short)0, c,
                                                 false, false);
}

static __device__ __forceinline__ v4bf pack4(float a, float b, float c, float d) {
  v4bf o;
  o[0] = (__bf16)a; o[1] = (__bf16)b; o[2] = (__bf16)c; o[3] = (__bf16)d;
  return o;
}

// ---------------------------------------------------------------------------
// K1/K4: Y[4096x512] = X[4096x512] @ W[512x512] + bias
//  mode 0: store bf16, head-major ((b*NH+h)*SEQ+i)*DH+d   (QKV path)
//  mode 1: store fp32, row-major row*HID+col              (output projection)
// Block: 256 thr (8 waves) -> 32x64 tile; wave (w>>2,w&3) -> 16x16 subtile.
// K-step 64 (2 WMMA/stage, 8 stages). All staging loops are fixed-trip and
// vectorized: float4 global loads, b64 LDS stores.
// ---------------------------------------------------------------------------
__global__ __launch_bounds__(256) void gemm_proj(
    const float* __restrict__ X, const float* __restrict__ W,
    const float* __restrict__ bias, __bf16* __restrict__ dst_bf,
    float* __restrict__ dst_f, int mode) {
  __shared__ __align__(16) __bf16 As[32 * 64];   // X tile, row-major (m,k)
  __shared__ __align__(16) __bf16 Bs[64 * 64];   // W tile, transposed (n,k)
  const int tid  = threadIdx.x;
  const int lane = tid & 31;
  const int wave = tid >> 5;
  const int half = lane >> 4;
  const int mn   = lane & 15;
  const int m_off = (wave >> 2) * 16;
  const int n_off = (wave & 3) * 16;
  const int m0 = blockIdx.x * 32;
  const int n0 = blockIdx.y * 64;

  v8f c = {};
  for (int k0 = 0; k0 < HID; k0 += 64) {
    // X tile 32x64: 512 float4 quads, 2 per thread (coalesced b128 loads)
#pragma unroll
    for (int it = 0; it < 2; ++it) {
      int task = tid + it * 256;
      int r = task >> 4, c4 = (task & 15) * 4;
      float4 xv = *(const float4*)&X[(size_t)(m0 + r) * HID + k0 + c4];
      *(v4bf*)&As[r * 64 + c4] = pack4(xv.x, xv.y, xv.z, xv.w);
    }
    // W tile 64k x 64n, stored transposed (n,k): 4 coalesced strided b32 loads
    // packed into one b64 LDS store. Consecutive threads -> consecutive n.
#pragma unroll
    for (int it = 0; it < 4; ++it) {
      int task = tid + it * 256;
      int n = task & 63, kq = (task >> 6) * 4;
      const float* wp = &W[(size_t)(k0 + kq) * HID + n0 + n];
      float w0 = wp[0], w1 = wp[HID], w2 = wp[2 * HID], w3 = wp[3 * HID];
      *(v4bf*)&Bs[n * 64 + kq] = pack4(w0, w1, w2, w3);
    }
    if (k0 + 64 < HID) {  // prefetch next K-tile (global_prefetch)
      __builtin_prefetch(&X[(size_t)(m0 + (tid >> 3)) * HID + k0 + 64], 0, 1);
      __builtin_prefetch(&W[(size_t)(k0 + 64 + (tid >> 2)) * HID + n0], 0, 1);
    }
    __syncthreads();
    const __bf16* ap = &As[(m_off + mn) * 64 + half * 8];
    const __bf16* bp = &Bs[(n_off + mn) * 64 + half * 8];
    c = wmma_bf16(ld_a_frag(ap), ld_b_frag(bp), c);
    c = wmma_bf16(ld_a_frag(ap + 32), ld_b_frag(bp + 32), c);
    __syncthreads();
  }

  const int col = n0 + n_off + mn;
  const float bv = bias[col];
#pragma unroll
  for (int r = 0; r < 8; ++r) {                  // D layout: m = r + 8*half, n = lane&15
    int row = m0 + m_off + r + 8 * half;
    float val = c[r] + bv;
    if (mode == 0) {
      int bb = row >> 11, ii = row & (SEQ - 1);  // row = b*SEQ + i
      int h = col >> 6, d = col & 63;            // col = h*DH + d
      dst_bf[(((size_t)bb * NH + h) * SEQ + ii) * DH + d] = (__bf16)val;
    } else {
      dst_f[(size_t)row * HID + col] = val;
    }
  }
}

// ---------------------------------------------------------------------------
// K2: relsum[(b*NH+h)*SEQ+i][p] = dot(q row + k row, rel_table[p])  (MAX_REL=3)
// Collapses attn2+attn3 (2 x 8.6 GF) into one fused 3-entry gather table
// (reference adds them before softmax, so the sum is all we need).
// ---------------------------------------------------------------------------
__global__ __launch_bounds__(256) void rel_proj(
    const __bf16* __restrict__ qb, const __bf16* __restrict__ kb,
    const float* __restrict__ rel_table, float* __restrict__ relsum) {
  int idx = blockIdx.x * 256 + threadIdx.x;      // (b*NH+h)*SEQ + i
  if (idx >= Bb * NH * SEQ) return;
  const __bf16* q = qb + (size_t)idx * DH;
  const __bf16* k = kb + (size_t)idx * DH;
#pragma unroll
  for (int p = 0; p < 3; ++p) {
    float s = 0.f;
#pragma unroll 8
    for (int d = 0; d < DH; ++d)
      s += ((float)q[d] + (float)k[d]) * rel_table[p * DH + d];
    relsum[(size_t)idx * 3 + p] = s;
  }
}

// ---------------------------------------------------------------------------
// K3: logits + single-pass exact softmax.
// One block = (b, h, 16-row i-tile). 128 KB fp32 logit block lives in LDS
// (320 KB/WGP makes this possible). 8 waves x 16 j-tiles x 2 WMMA, then a
// vectorized (float4) softmax: 16 threads/row, each owning 128 contiguous
// columns -> ds_*_b128 + global_store_b128 for the (dominant) attn write.
// ---------------------------------------------------------------------------
__global__ __launch_bounds__(256) void attn_softmax(
    const __bf16* __restrict__ qb, const __bf16* __restrict__ kb,
    const float* __restrict__ relsum, const int* __restrict__ pos,
    const unsigned char* __restrict__ mask, float* __restrict__ attn) {
  extern __shared__ float lds[];
  float* logits  = lds;                 // 16*SEQ = 32768 floats (128 KB)
  float* partial = lds + 16 * SEQ;      // 256
  float* rowval  = partial + 256;       // 16

  const int tid = threadIdx.x, lane = tid & 31, wave = tid >> 5;
  const int half = lane >> 4, mn = lane & 15;
  const int i0 = blockIdx.x * 16;
  const int h = blockIdx.y, bz = blockIdx.z;
  const size_t hb = ((size_t)bz * NH + h) * SEQ;

  // Q A-frags for this 16-row tile (reused across all 128 j-tiles)
  const __bf16* qrow = qb + (hb + i0 + mn) * DH;
  v16bf a0 = ld_a_frag(qrow + half * 8);        // d = 0..31
  v16bf a1 = ld_a_frag(qrow + 32 + half * 8);   // d = 32..63

  for (int jt = wave; jt < SEQ / 16; jt += 8) {
    const int j0 = jt * 16;
    // K rows load directly as B-frags (row-major k == B = k^T, contiguous K run)
    const __bf16* krow = kb + (hb + j0 + mn) * DH;
    v16bf b0 = ld_b_frag(krow + half * 8);
    v16bf b1 = ld_b_frag(krow + 32 + half * 8);
    v8f c = {};
    c = wmma_bf16(a0, b0, c);
    c = wmma_bf16(a1, b1, c);
#pragma unroll
    for (int r = 0; r < 8; ++r) {
      int ml = r + 8 * half;
      int i = i0 + ml, j = j0 + mn;
      int p = pos[(size_t)i * SEQ + j];
      float val = (c[r] + relsum[(hb + i) * 3 + p]) * 0.125f;
      if (mask[((size_t)bz * SEQ + i) * SEQ + j]) val = -1.0e9f;
      logits[ml * SEQ + j] = val;
    }
  }
  __syncthreads();

  // Softmax: 16 threads per row; thread owns 128 contiguous cols (b128 ops).
  const int rr = tid >> 4, sub = tid & 15;
  float* lrow = logits + rr * SEQ + sub * 128;
  float mx = -3.4e38f;
#pragma unroll 8
  for (int kq = 0; kq < 32; ++kq) {
    float4 v = ((const float4*)lrow)[kq];
    mx = fmaxf(mx, fmaxf(fmaxf(v.x, v.y), fmaxf(v.z, v.w)));
  }
  partial[tid] = mx;
  __syncthreads();
  if (tid < 16) {
    float v = partial[tid * 16];
    for (int t = 1; t < 16; ++t) v = fmaxf(v, partial[tid * 16 + t]);
    rowval[tid] = v;
  }
  __syncthreads();
  const float rmax = rowval[rr];
  float sum = 0.f;
#pragma unroll 4
  for (int kq = 0; kq < 32; ++kq) {
    float4 v = ((const float4*)lrow)[kq];
    v.x = __expf(v.x - rmax); v.y = __expf(v.y - rmax);
    v.z = __expf(v.z - rmax); v.w = __expf(v.w - rmax);
    sum += (v.x + v.y) + (v.z + v.w);
    ((float4*)lrow)[kq] = v;
  }
  partial[tid] = sum;
  __syncthreads();
  if (tid < 16) {
    float v = 0.f;
    for (int t = 0; t < 16; ++t) v += partial[tid * 16 + t];
    rowval[tid] = v;
  }
  __syncthreads();
  const float inv = 1.0f / rowval[rr];
  float* orow = attn + (hb + i0 + rr) * SEQ + sub * 128;  // final attn, b128 stores
#pragma unroll 4
  for (int kq = 0; kq < 32; ++kq) {
    float4 v = ((const float4*)lrow)[kq];
    v.x *= inv; v.y *= inv; v.z *= inv; v.w *= inv;
    ((float4*)orow)[kq] = v;
  }
}

// ---------------------------------------------------------------------------
// K5: x[b,i,h*64+d] = sum_j attn[b,h,i,j] * v[b,h,j,d]
// Block = (b,h,16 i-rows), 4 waves cover d0 = 0/16/32/48.
// K-step 64 (2 WMMA/stage, 32 stages -> 64 WMMA/wave).
// ---------------------------------------------------------------------------
__global__ __launch_bounds__(128) void attn_v(
    const float* __restrict__ attn, const __bf16* __restrict__ vb,
    float* __restrict__ x) {
  __shared__ __align__(16) __bf16 As[16 * 64];  // attn tile bf16, row-major (i,j)
  __shared__ __align__(16) __bf16 Vs[64 * 64];  // v tile transposed (d,j)
  const int tid = threadIdx.x, lane = tid & 31, wave = tid >> 5;
  const int half = lane >> 4, mn = lane & 15;
  const int d0 = wave * 16;
  const int i0 = blockIdx.x * 16;
  const int h = blockIdx.y, bz = blockIdx.z;
  const size_t hb = ((size_t)bz * NH + h) * SEQ;
  const float* arow = attn + (hb + i0) * SEQ;

  v8f c = {};
  for (int j0 = 0; j0 < SEQ; j0 += 64) {
    // attn tile 16x64 fp32 -> bf16: 256 quads, 2 per thread (b128 loads)
#pragma unroll
    for (int it = 0; it < 2; ++it) {
      int task = tid + it * 128;
      int r = task >> 4, c4 = (task & 15) * 4;
      float4 av = *(const float4*)&arow[(size_t)r * SEQ + j0 + c4];
      *(v4bf*)&As[r * 64 + c4] = pack4(av.x, av.y, av.z, av.w);
    }
    // v tile 64j x 64d, stored transposed (d,j): 4 coalesced strided b16 loads
    // packed into one b64 LDS store. Consecutive threads -> consecutive d.
#pragma unroll
    for (int it = 0; it < 8; ++it) {
      int task = tid + it * 128;
      int d = task & 63, jq = (task >> 6) * 4;
      const __bf16* vp = &vb[(hb + j0 + jq) * DH + d];
      v4bf o;
      o[0] = vp[0]; o[1] = vp[DH]; o[2] = vp[2 * DH]; o[3] = vp[3 * DH];
      *(v4bf*)&Vs[d * 64 + jq] = o;
    }
    if (j0 + 64 < SEQ) {  // prefetch next K-tile
      __builtin_prefetch(&arow[(size_t)(tid >> 3) * SEQ + j0 + 64], 0, 1);
      __builtin_prefetch(&vb[(hb + j0 + 64 + (tid >> 1)) * DH], 0, 1);
    }
    __syncthreads();
    const __bf16* ap = &As[mn * 64 + half * 8];
    const __bf16* bp = &Vs[(d0 + mn) * 64 + half * 8];
    c = wmma_bf16(ld_a_frag(ap), ld_b_frag(bp), c);
    c = wmma_bf16(ld_a_frag(ap + 32), ld_b_frag(bp + 32), c);
    __syncthreads();
  }
#pragma unroll
  for (int r = 0; r < 8; ++r) {
    int row = i0 + r + 8 * half;
    x[((size_t)bz * SEQ + row) * HID + h * DH + d0 + mn] = c[r];
  }
}

// ---------------------------------------------------------------------------
// Launch. Workspace layout (needs ~21 MB):
//  [0,4M) qb bf16 | [4M,8M) kb | [8M,12M) vb | [12M,+384K) relsum |
//  [13M,21M) x fp32
// d_out = out (2*2048*512 f32) ++ attn (2*8*2048*2048 f32)
// ---------------------------------------------------------------------------
extern "C" void kernel_launch(void* const* d_in, const int* in_sizes, int n_in,
                              void* d_out, int out_size, void* d_ws,
                              size_t ws_size, hipStream_t stream) {
  (void)in_sizes; (void)n_in; (void)out_size; (void)ws_size;
  const float* query = (const float*)d_in[0];
  const float* key_  = (const float*)d_in[1];
  const float* value = (const float*)d_in[2];
  const unsigned char* mask = (const unsigned char*)d_in[3];
  const int* pos = (const int*)d_in[4];
  const float* Wq = (const float*)d_in[5];
  const float* bq = (const float*)d_in[6];
  const float* Wk = (const float*)d_in[7];
  const float* bk = (const float*)d_in[8];
  const float* Wv = (const float*)d_in[9];
  const float* bv = (const float*)d_in[10];
  const float* Wo = (const float*)d_in[11];
  const float* bo = (const float*)d_in[12];
  const float* rel_table = (const float*)d_in[13];

  char* ws = (char*)d_ws;
  __bf16* qb    = (__bf16*)(ws);
  __bf16* kb    = (__bf16*)(ws + (4ull << 20));
  __bf16* vb    = (__bf16*)(ws + (8ull << 20));
  float* relsum = (float*)(ws + (12ull << 20));
  float* xws    = (float*)(ws + (13ull << 20));

  float* out  = (float*)d_out;
  float* attn = out + (size_t)Bb * SEQ * HID;

  dim3 blk(256);
  dim3 gG((Bb * SEQ) / 32, HID / 64);  // 128 x 8
  gemm_proj<<<gG, blk, 0, stream>>>(query, Wq, bq, qb, nullptr, 0);
  gemm_proj<<<gG, blk, 0, stream>>>(key_,  Wk, bk, kb, nullptr, 0);
  gemm_proj<<<gG, blk, 0, stream>>>(value, Wv, bv, vb, nullptr, 0);
  rel_proj<<<dim3((Bb * NH * SEQ) / 256), blk, 0, stream>>>(qb, kb, rel_table,
                                                            relsum);
  size_t smem = (16 * SEQ + 256 + 16) * sizeof(float);  // ~129.1 KB of 320 KB LDS
  (void)hipFuncSetAttribute((const void*)attn_softmax,
                            hipFuncAttributeMaxDynamicSharedMemorySize,
                            (int)smem);
  attn_softmax<<<dim3(SEQ / 16, NH, Bb), blk, smem, stream>>>(qb, kb, relsum,
                                                              pos, mask, attn);
  attn_v<<<dim3(SEQ / 16, NH, Bb), dim3(128), 0, stream>>>(attn, vb, xws);
  gemm_proj<<<gG, blk, 0, stream>>>(xws, Wo, bo, nullptr, out, 1);
}
